// Seq_Encoder_8194797600815
// MI455X (gfx1250) — matrix operand order repaired and checked by reading, not compiled, not run on hardware
//
#include <hip/hip_runtime.h>
#include <hip/hip_bf16.h>

// ---------- types ----------
typedef __attribute__((ext_vector_type(16))) __bf16 v16bf;
typedef __attribute__((ext_vector_type(8)))  __bf16 v8bf;
typedef __attribute__((ext_vector_type(8)))  float  v8f;

#define M_ROWS 4096   // B_SZ * L_SEQ
#define L_SEQ_  1024

// ---------- WMMA fragment loads ----------
// A: row-major (M x K) bf16.  lane = M row (lanes 0-15 & 16-31 both M=0..15),
// elements 0..7 = K (k0 + hi*8 .. +7), elements 8..15 = K (k0 + 16 + hi*8 ..).
__device__ __forceinline__ v16bf load_fragA(const __bf16* Abase, int lda, int k0) {
  int lane = threadIdx.x & 31;
  int r  = lane & 15;
  int hi = (lane >> 4) & 1;
  const __bf16* p = Abase + (size_t)r * lda + k0 + hi * 8;
  __builtin_prefetch(p + 64, 0, 1);            // gfx1250 global prefetch path
  v8bf lo = *(const v8bf*)(p);
  v8bf hh = *(const v8bf*)(p + 16);
  v16bf f;
#pragma unroll
  for (int i = 0; i < 8; ++i) { f[i] = lo[i]; f[i + 8] = hh[i]; }
  return f;
}

// B: row-major (K x Npad) bf16. lane = N col (n0 + lane%16), elements 0..7 = K (k0 + hi*8 ..), 8..15 = K (k0 + 16 + hi*8 ..).
__device__ __forceinline__ v16bf load_fragB(const __bf16* B, int ldb, int n0, int k0) {
  int lane = threadIdx.x & 31;
  int hi = (lane >> 4) & 1;
  const __bf16* p = B + (size_t)(k0 + hi * 8) * ldb + n0 + (lane & 15);
  v16bf f;
#pragma unroll
  for (int i = 0; i < 8; ++i) { f[i] = p[(size_t)i * ldb]; f[i + 8] = p[(size_t)(i + 16) * ldb]; }
  return f;
}

#define WMMA_BF16(a, b, c) __builtin_amdgcn_wmma_f32_16x16x32_bf16( \
    false, (a), false, (b), (short)0, (c), false, false)

// ---------- generic bf16 WMMA GEMM:  C = act(A @ B + bias) ----------
// A: (M x K) bf16 compact, B: (K x Npad64) bf16, cols >= N are zero,
// Npad64 = round_up(N, 64) == ldb.  Each wave computes a 16x64 strip
// (4 accumulators reuse one A fragment -> 2.5 loads/WMMA instead of 4).
template <bool BF16OUT>
__global__ __launch_bounds__(256) void gemm_wmma(
    const __bf16* __restrict__ A, const __bf16* __restrict__ B, void* __restrict__ Cout,
    int M, int N, int K, int ldb, int ldc, int coff,
    const float* __restrict__ bias, int relu) {
  int wave = threadIdx.x >> 5;
  int mtiles  = M >> 4;
  int nstrips = ldb >> 6;
  int wid = blockIdx.x * 8 + wave;
  if (wid >= mtiles * nstrips) return;     // wave-uniform: safe before WMMA
  int m0 = (wid / nstrips) << 4;
  int n0 = (wid % nstrips) << 6;

  v8f acc[4] = {};
  const __bf16* Abase = A + (size_t)m0 * K;
  for (int k0 = 0; k0 < K; k0 += 32) {
    v16bf a  = load_fragA(Abase, K, k0);
    v16bf b0 = load_fragB(B, ldb, n0,      k0);
    v16bf b1 = load_fragB(B, ldb, n0 + 16, k0);
    v16bf b2 = load_fragB(B, ldb, n0 + 32, k0);
    v16bf b3 = load_fragB(B, ldb, n0 + 48, k0);
    acc[0] = WMMA_BF16(a, b0, acc[0]);
    acc[1] = WMMA_BF16(a, b1, acc[1]);
    acc[2] = WMMA_BF16(a, b2, acc[2]);
    acc[3] = WMMA_BF16(a, b3, acc[3]);
  }

  int lane = threadIdx.x & 31;
  int nl = lane & 15;
  int hi = (lane >> 4) & 1;
#pragma unroll
  for (int t = 0; t < 4; ++t) {
    int n = n0 + t * 16 + nl;
    if (n >= N) continue;                  // post-WMMA divergence: fine
    float bv = bias ? bias[n] : 0.f;
#pragma unroll
    for (int r = 0; r < 8; ++r) {
      int m = m0 + r + hi * 8;
      float v = acc[t][r] + bv;
      if (relu) v = fmaxf(v, 0.f);
      if (BF16OUT) ((__bf16*)Cout)[(size_t)m * ldc + coff + n] = (__bf16)v;
      else         ((float*) Cout)[(size_t)m * ldc + coff + n] = v;
    }
  }
}

// ---------- weight convert (f32 (K x N) -> bf16 (K x Npad64), zero-pad) ----------
__global__ void cvt_pad(const float* __restrict__ W, __bf16* __restrict__ Bb,
                        int K, int N, int Npad) {
  int idx = blockIdx.x * blockDim.x + threadIdx.x;
  if (idx >= K * Npad) return;
  int k = idx / Npad, n = idx % Npad;
  Bb[idx] = (__bf16)((n < N) ? W[(size_t)k * N + n] : 0.f);
}

// ---------- im2row patches for conv bank: X7[row][t*128+i] = embed[ids[l+t-3]][i] ----------
__global__ void build_x7(const int* __restrict__ ids, const float* __restrict__ embed,
                         __bf16* __restrict__ X7) {
  size_t idx = (size_t)blockIdx.x * 256 + threadIdx.x;
  if (idx >= (size_t)M_ROWS * 896) return;
  int row = (int)(idx / 896), col = (int)(idx % 896);
  int t = col / 128, i = col % 128;
  int b = row >> 10, l = row & 1023;
  int ls = l + t - 3;
  float v = 0.f;
  if (ls >= 0 && ls < L_SEQ_) v = embed[(size_t)ids[b * L_SEQ_ + ls] * 128 + i];
  X7[idx] = (__bf16)v;
}

// ---------- combined conv weight (896 x 256) + combined bias ----------
__global__ void build_wc(const float* w0, const float* w1, const float* w2, const float* w3,
                         const float* b0, const float* b1, const float* b2, const float* b3,
                         __bf16* __restrict__ Wc, float* __restrict__ biasC) {
  int idx = blockIdx.x * blockDim.x + threadIdx.x;   // 896*256
  if (idx >= 896 * 256) return;
  int r = idx / 256, c = idx % 256;
  int t = r / 128, i = r % 128;
  const float* w; int cl, k;
  if (c < 32)       { w = w0; cl = c;       k = 1; }
  else if (c < 64)  { w = w1; cl = c - 32;  k = 3; }
  else if (c < 128) { w = w2; cl = c - 64;  k = 5; }
  else              { w = w3; cl = c - 128; k = 7; }
  int off = (7 - k) / 2;
  float v = 0.f;
  if (t >= off && t < off + k) v = w[(size_t)cl * 128 * k + i * k + (t - off)];
  Wc[idx] = (__bf16)v;
  if (idx < 256) {
    float bb = (idx < 32)  ? b0[idx]
             : (idx < 64)  ? b1[idx - 32]
             : (idx < 128) ? b2[idx - 64]
             :               b3[idx - 128];
    biasC[idx] = bb;
  }
}

// ---------- dt = softplus(zx[:,640+h]+bias), dA = exp(dt * -exp(A_log)) ----------
__global__ void dt_kernel(const float* __restrict__ zx, const float* __restrict__ dt_bias,
                          const float* __restrict__ A_log,
                          float* __restrict__ dt, float* __restrict__ dA) {
  int idx = blockIdx.x * blockDim.x + threadIdx.x;   // 4096*4
  if (idx >= M_ROWS * 4) return;
  int row = idx >> 2, h = idx & 3;
  float v = zx[(size_t)row * 656 + 640 + h] + dt_bias[h];
  float sp = (v > 20.f) ? v : log1pf(expf(v));
  float a = -expf(A_log[h]);
  dt[idx] = sp;
  dA[idx] = expf(sp * a);
}

// ---------- causal depthwise conv (D_CONV=4) + SiLU over xBC (384 ch) ----------
__global__ void dwconv_kernel(const float* __restrict__ zx, const float* __restrict__ conv_w,
                              const float* __restrict__ conv_b, float* __restrict__ xBCs) {
  int idx = blockIdx.x * blockDim.x + threadIdx.x;   // 4096*384
  if (idx >= M_ROWS * 384) return;
  int row = idx / 384, c = idx % 384;
  int l = row & 1023;
  float acc = conv_b[c];
#pragma unroll
  for (int j = 0; j < 4; ++j) {
    int ls = l - 3 + j;
    if (ls >= 0) acc += conv_w[c * 4 + j] * zx[(size_t)(row - 3 + j) * 656 + 256 + c];
  }
  xBCs[idx] = acc / (1.f + expf(-acc));              // SiLU
}

// ---------- SSM scan: one block per (batch,head); 64x64 state in registers ----------
__global__ __launch_bounds__(256) void scan_kernel(
    const float* __restrict__ xBCs, const float* __restrict__ dt,
    const float* __restrict__ dA, const float* __restrict__ Dp,
    float* __restrict__ y) {
  __shared__ float sx[64][64], sB[64][64], sC[64][64];
  __shared__ float sdt[64], sdA[64];
  int bh = blockIdx.x; int b = bh >> 2, h = bh & 3;
  int tid = threadIdx.x;
  int p = tid >> 2, q = tid & 3, nb = q * 16;        // thread owns n in [nb, nb+16)
  float Dh = Dp[h];
  float hst[16];
#pragma unroll
  for (int j = 0; j < 16; ++j) hst[j] = 0.f;
  int rbase = b * L_SEQ_;
  for (int chunk = 0; chunk < 16; ++chunk) {
    int t0 = chunk * 64;
#pragma unroll
    for (int j = 0; j < 16; ++j) {                   // stage 64 steps of x/B/C
      int e = tid + 256 * j;
      int tt = e >> 6, pp = e & 63;
      int row = rbase + t0 + tt;
      sx[tt][pp] = xBCs[(size_t)row * 384 + h * 64 + pp];
      sB[tt][pp] = xBCs[(size_t)row * 384 + 256 + pp];
      sC[tt][pp] = xBCs[(size_t)row * 384 + 320 + pp];
    }
    if (tid < 64) {
      int row = rbase + t0 + tid;
      sdt[tid] = dt[row * 4 + h];
      sdA[tid] = dA[row * 4 + h];
    }
    __syncthreads();
    for (int tt = 0; tt < 64; ++tt) {
      float a = sdA[tt], d = sdt[tt];
      float xp = sx[tt][p];
      float coef = d * xp;
      float acc = 0.f;
#pragma unroll
      for (int j = 0; j < 16; ++j) {
        hst[j] = hst[j] * a + coef * sB[tt][nb + j];
        acc += hst[j] * sC[tt][nb + j];
      }
      acc += __shfl_xor(acc, 1, 32);                 // reduce over q (wave32 lanes)
      acc += __shfl_xor(acc, 2, 32);
      if (q == 0) {
        int row = rbase + t0 + tt;
        y[(size_t)row * 256 + h * 64 + p] = acc + Dh * xp;
      }
    }
    __syncthreads();
  }
}

// ---------- gating + RMS norm -> bf16 ----------
__global__ __launch_bounds__(256) void gate_norm_kernel(
    const float* __restrict__ y, const float* __restrict__ zx,
    const float* __restrict__ norm_w, __bf16* __restrict__ yz) {
  __shared__ float red[8];
  int row = blockIdx.x, c = threadIdx.x;
  float yv = y[(size_t)row * 256 + c];
  float zv = zx[(size_t)row * 656 + c];
  float v = yv * (zv / (1.f + expf(-zv)));
  float s = v * v;
#pragma unroll
  for (int off = 16; off > 0; off >>= 1) s += __shfl_xor(s, off, 32);
  if ((c & 31) == 0) red[c >> 5] = s;
  __syncthreads();
  float tot = 0.f;
#pragma unroll
  for (int i = 0; i < 8; ++i) tot += red[i];
  float scale = rsqrtf(tot / 256.f + 1e-5f);
  yz[(size_t)row * 256 + c] = (__bf16)(v * scale * norm_w[c]);
}

// ---------- host helpers ----------
static inline char* wsalloc(char*& p, size_t bytes) {
  char* r = p;
  p += (bytes + 255) & ~(size_t)255;
  return r;
}

static void gemm(hipStream_t s, const __bf16* A, const __bf16* B, void* C,
                 int M, int N, int K, int ldc, int coff,
                 const float* bias, int relu, bool bf16out) {
  int npad64 = (N + 63) & ~63;             // == ldb of the padded B buffer
  int work = (M >> 4) * (npad64 >> 6);     // mtiles * nstrips
  dim3 grid((work + 7) / 8);
  if (bf16out)
    gemm_wmma<true><<<grid, 256, 0, s>>>(A, B, C, M, N, K, npad64, ldc, coff, bias, relu);
  else
    gemm_wmma<false><<<grid, 256, 0, s>>>(A, B, C, M, N, K, npad64, ldc, coff, bias, relu);
}

extern "C" void kernel_launch(void* const* d_in, const int* in_sizes, int n_in,
                              void* d_out, int out_size, void* d_ws, size_t ws_size,
                              hipStream_t stream) {
  (void)in_sizes; (void)n_in; (void)out_size; (void)ws_size;
  // input order (setup_inputs insertion order, nested dicts in insertion order):
  // 0 seq_input, 1 poc_input,
  // 2..12  seq_emb:  embed, c0w,c0b, c1w,c1b, c2w,c2b, c3w,c3b, proj_w, proj_b
  // 13..23 poc_emb:  same
  // 24..31 m_glo:    in_proj_w, conv_w, conv_b, dt_bias, A_log, D, norm_w, out_proj_w
  // 32..39 m_loc:    same
  // 40 lin_w, 41 lin_b

  char* wp = (char*)d_ws;
  __bf16* gl    = (__bf16*)wsalloc(wp, (size_t)M_ROWS * 256 * 2);   // concat[g,l] bf16
  __bf16* linB  = (__bf16*)wsalloc(wp, (size_t)256 * 64 * 2);
  __bf16* X7    = (__bf16*)wsalloc(wp, (size_t)M_ROWS * 896 * 2);
  __bf16* Wc    = (__bf16*)wsalloc(wp, (size_t)896 * 256 * 2);
  float*  biasC = (float*) wsalloc(wp, 256 * 4);
  __bf16* projB = (__bf16*)wsalloc(wp, (size_t)256 * 128 * 2);
  __bf16* inB   = (__bf16*)wsalloc(wp, (size_t)128 * 704 * 2);      // 644 -> pad 704
  __bf16* outB  = (__bf16*)wsalloc(wp, (size_t)256 * 128 * 2);
  __bf16* hconv = (__bf16*)wsalloc(wp, (size_t)M_ROWS * 256 * 2);
  __bf16* ubuf  = (__bf16*)wsalloc(wp, (size_t)M_ROWS * 128 * 2);
  float*  zx    = (float*) wsalloc(wp, (size_t)M_ROWS * 656 * 4);
  float*  xBCs  = (float*) wsalloc(wp, (size_t)M_ROWS * 384 * 4);
  float*  dtb   = (float*) wsalloc(wp, (size_t)M_ROWS * 4 * 4);
  float*  dAb   = (float*) wsalloc(wp, (size_t)M_ROWS * 4 * 4);
  float*  ybuf  = (float*) wsalloc(wp, (size_t)M_ROWS * 256 * 4);
  __bf16* yz    = (__bf16*)wsalloc(wp, (size_t)M_ROWS * 256 * 2);

  cvt_pad<<<(256 * 64 + 255) / 256, 256, 0, stream>>>((const float*)d_in[40], linB, 256, 64, 64);

  for (int br = 0; br < 2; ++br) {
    const int* ids = (const int*)d_in[br ? 1 : 0];
    int eb = br ? 13 : 2;
    int mb = br ? 32 : 24;
    const float* embed   = (const float*)d_in[eb + 0];
    const float* c0w = (const float*)d_in[eb + 1]; const float* c0b = (const float*)d_in[eb + 2];
    const float* c1w = (const float*)d_in[eb + 3]; const float* c1b = (const float*)d_in[eb + 4];
    const float* c2w = (const float*)d_in[eb + 5]; const float* c2b = (const float*)d_in[eb + 6];
    const float* c3w = (const float*)d_in[eb + 7]; const float* c3b = (const float*)d_in[eb + 8];
    const float* proj_w = (const float*)d_in[eb + 9];
    const float* proj_b = (const float*)d_in[eb + 10];
    const float* in_proj_w  = (const float*)d_in[mb + 0];
    const float* conv_w     = (const float*)d_in[mb + 1];
    const float* conv_b     = (const float*)d_in[mb + 2];
    const float* dt_bias    = (const float*)d_in[mb + 3];
    const float* A_log      = (const float*)d_in[mb + 4];
    const float* Dp         = (const float*)d_in[mb + 5];
    const float* norm_w     = (const float*)d_in[mb + 6];
    const float* out_proj_w = (const float*)d_in[mb + 7];

    build_x7<<<(M_ROWS * 896 + 255) / 256, 256, 0, stream>>>(ids, embed, X7);
    build_wc<<<(896 * 256 + 255) / 256, 256, 0, stream>>>(c0w, c1w, c2w, c3w,
                                                          c0b, c1b, c2b, c3b, Wc, biasC);
    cvt_pad<<<(256 * 128 + 255) / 256, 256, 0, stream>>>(proj_w, projB, 256, 128, 128);
    cvt_pad<<<(128 * 704 + 255) / 256, 256, 0, stream>>>(in_proj_w, inB, 128, 644, 704);
    cvt_pad<<<(256 * 128 + 255) / 256, 256, 0, stream>>>(out_proj_w, outB, 256, 128, 128);

    // conv bank as GEMM: relu(X7 @ Wc + biasC) -> hconv bf16 (4096 x 256)
    gemm(stream, X7, Wc, hconv, M_ROWS, 256, 896, 256, 0, biasC, 1, true);
    // proj: relu(hconv @ proj_w + proj_b) -> u bf16 (4096 x 128)
    gemm(stream, hconv, projB, ubuf, M_ROWS, 128, 256, 128, 0, proj_b, 1, true);
    // in_proj: u @ in_proj_w -> zx f32 (4096 x 656, cols 644.. zero)
    gemm(stream, ubuf, inB, zx, M_ROWS, 656, 128, 656, 0, nullptr, 0, false);

    dt_kernel<<<(M_ROWS * 4 + 255) / 256, 256, 0, stream>>>(zx, dt_bias, A_log, dtb, dAb);
    dwconv_kernel<<<(M_ROWS * 384 + 255) / 256, 256, 0, stream>>>(zx, conv_w, conv_b, xBCs);
    scan_kernel<<<16, 256, 0, stream>>>(xBCs, dtb, dAb, Dp, ybuf);
    gate_norm_kernel<<<M_ROWS, 256, 0, stream>>>(ybuf, zx, norm_w, yz);

    // out_proj -> write branch slice of concat buffer (bf16, ldc=256, col offset br*128)
    gemm(stream, yz, outB, gl, M_ROWS, 128, 256, 256, br * 128, nullptr, 0, true);
  }

  // final: relu(concat @ lin_w + lin_b) -> f32 d_out (4096 x 64)
  gemm(stream, gl, linB, d_out, M_ROWS, 64, 256, 64, 0, (const float*)d_in[41], 1, false);
}
